// DenoisedSasrec_4647154615564
// MI455X (gfx1250) — matrix operand order repaired and checked by reading, not compiled, not run on hardware
//
#include <hip/hip_runtime.h>
#include <hip/hip_bf16.h>

typedef unsigned short u16;
typedef __attribute__((ext_vector_type(16))) __bf16 v16bf;
typedef __attribute__((ext_vector_type(8)))  float  v8f;

// Sizes are compile-time constants from the reference: B=64, L=H=512.
#define LDIM 512
#define BDIM 64
#define MAT  (512*512)          // per-batch matrix elements
#define KBLK 64                 // K slab staged per LDS buffer
#define LPAD 72                 // padded LDS row stride (bank-conflict avoidance)

static __device__ __forceinline__ u16 f2bf(float f) {
  union { float f; unsigned u; } x; x.f = f;
  unsigned r = x.u + 0x7FFFu + ((x.u >> 16) & 1u);   // round-to-nearest-even
  return (u16)(r >> 16);
}

static __device__ __forceinline__ float fsilu(float x) {
  return x * (1.0f / (1.0f + __expf(-x)));
}

// CDNA5 async global->LDS copy, 16B per lane (ISA 15.18.3 op 98, ASYNCcnt).
// Chunk displacement goes into the 24-bit instruction offset (no VALU addressing).
template<int IMM>
static __device__ __forceinline__ void async_b128(unsigned lds_addr, unsigned long long gaddr) {
  asm volatile("global_load_async_to_lds_b128 %0, %1, off offset:%c2"
               :: "v"(lds_addr), "v"(gaddr), "n"(IMM) : "memory");
}
static __device__ __forceinline__ void wait_async9() {
  asm volatile("s_wait_asynccnt 0x9" ::: "memory");
}
static __device__ __forceinline__ void wait_async0() {
  asm volatile("s_wait_asynccnt 0x0" ::: "memory");
}

// A-fragment (16x32 bf16, ISA 7.12.2): lane = (hi,m); hi=0 -> K 0..7,16..23 ; hi=1 -> K 8..15,24..31
static __device__ __forceinline__ v16bf load_fragA(const u16* row, int k0, int hi) {
  union { v16bf v; uint4 q[2]; } u;
  u.q[0] = *reinterpret_cast<const uint4*>(row + k0 + 8 * hi);
  u.q[1] = *reinterpret_cast<const uint4*>(row + k0 + 16 + 8 * hi);
  return u.v;
}

// B-fragment (32x16 bf16): lane = (hi,n); hi=0 -> K 0..7,16..23 ; hi=1 -> K 8..15,24..31
static __device__ __forceinline__ v16bf load_fragB(const u16* row, int k0, int hi) {
  union { v16bf v; uint4 q[2]; } u;
  u.q[0] = *reinterpret_cast<const uint4*>(row + k0 + 8 * hi);
  u.q[1] = *reinterpret_cast<const uint4*>(row + k0 + 16 + 8 * hi);
  return u.v;
}

// Epilogues:
// 0: silu -> bf16, row-major [*,512]
// 1: silu -> bf16, v-transposed store: col=(b*512+l), row=h  ->  Vt[b][h][l]
// 2: gamma/beta -> bf16 row-major
// 3: attention mask + gumbel-sigmoid sparse mask + relu^2 -> bf16 (per-batch z)
// 4: * 1/(L*H) -> f32 output (per-batch z)
template<int EPI>
static __device__ __forceinline__ void epi_store(
    float v, int row, int col, int z, size_t cZ,
    u16* __restrict__ Cb, float* __restrict__ Cf,
    const float* __restrict__ gvec, const float* __restrict__ bvec,
    const int* __restrict__ maski, const float* __restrict__ smask) {
  if (EPI == 0) {
    Cb[(size_t)row * LDIM + col] = f2bf(fsilu(v));
  } else if (EPI == 1) {
    Cb[(((size_t)(col >> 9)) * LDIM + row) * LDIM + (col & 511)] = f2bf(fsilu(v));
  } else if (EPI == 2) {
    Cb[(size_t)row * LDIM + col] = f2bf(v * gvec[col] + bvec[col]);
  } else if (EPI == 3) {
    // merged mask: pad(col) | future(row<col), diagonal always kept
    bool masked = (row != col) && ((maski[(size_t)z * LDIM + col] == 0) || (row < col));
    float t = masked ? 0.0f : v;
    t *= smask[(size_t)row * LDIM + col];
    t = (t > 0.0f) ? t * t : 0.0f;              // relu^2 ; /(L*H) deferred to EPI 4
    Cb[(size_t)z * cZ + (size_t)row * LDIM + col] = f2bf(t);
  } else {
    Cf[(size_t)z * cZ + (size_t)row * LDIM + col] = v * (1.0f / 262144.0f);
  }
}

#define WMMA_BF16(ACC, AF, BF) \
  ACC = __builtin_amdgcn_wmma_f32_16x16x32_bf16(false, AF, false, BF, (short)0, ACC, false, false)

// C[M,N] = epi( A[M,512] @ Bt[N,512]^T )
// Block tile 32x256 (8 waves, wave tile 16x64). K staged through double-buffered
// LDS slabs of K=64 filled with global_load_async_to_lds_b128 (ASYNCcnt),
// fragments pulled from LDS with ds_load_b128. Fully unrolled over the 8 K-blocks
// so all staging addresses are base + immediate offset.
template<int EPI>
__global__ __launch_bounds__(256) void gemm_k(
    const u16* __restrict__ A, const u16* __restrict__ Bt,
    size_t aZ, size_t bZ, size_t cZ,
    u16* __restrict__ Cb, float* __restrict__ Cf,
    const float* __restrict__ gvec, const float* __restrict__ bvec,
    const int* __restrict__ maski, const float* __restrict__ smask) {
  __shared__ __align__(16) u16 As[2][32][LPAD];
  __shared__ __align__(16) u16 Bs[2][256][LPAD];

  const int t    = threadIdx.x;
  const int w    = t >> 5;
  const int lane = t & 31;
  const int hi   = lane >> 4;
  const int ln   = lane & 15;
  const int z    = blockIdx.z;
  const int mrowB = blockIdx.y * 32;
  const int ncolB = blockIdx.x * 256;

  // Per-thread copy assignment: B row t (8x16B chunks) + 1 A chunk.
  const int ar = t >> 3, ac = t & 7;            // A: 32 rows x 8 chunks
  const unsigned long long gBrow =
      (unsigned long long)(Bt + (size_t)z * bZ + (size_t)(ncolB + t) * LDIM);
  const unsigned long long gArow =
      (unsigned long long)(A + (size_t)z * aZ + (size_t)(mrowB + ar) * LDIM + ac * 8);
  const unsigned lB0 = (unsigned)(size_t)&Bs[0][t][0];
  const unsigned lB1 = (unsigned)(size_t)&Bs[1][t][0];
  const unsigned lA0 = (unsigned)(size_t)&As[0][ar][ac * 8];
  const unsigned lA1 = (unsigned)(size_t)&As[1][ar][ac * 8];

  v8f acc0 = {}, acc1 = {}, acc2 = {}, acc3 = {};

  const u16* aRow  = &As[0][(w & 1) * 16 + ln][0];
  const u16* bRow0 = &Bs[0][(w >> 1) * 64 + 0 * 16 + ln][0];
  const u16* bRow1 = &Bs[0][(w >> 1) * 64 + 1 * 16 + ln][0];
  const u16* bRow2 = &Bs[0][(w >> 1) * 64 + 2 * 16 + ln][0];
  const u16* bRow3 = &Bs[0][(w >> 1) * 64 + 3 * 16 + ln][0];
  const size_t bufA = (size_t)(&As[1][0][0] - &As[0][0][0]);   // elems between buffers
  const size_t bufB = (size_t)(&Bs[1][0][0] - &Bs[0][0][0]);

  // stage K-block KB (compile-time) into buffer BF; all offsets are immediates
  #define STAGE(KB, BF)                                                          \
    do {                                                                         \
      const unsigned _lb = (BF) ? lB1 : lB0;                                     \
      const unsigned _la = (BF) ? lA1 : lA0;                                     \
      async_b128<(KB)*128 +   0>(_lb +   0, gBrow);                              \
      async_b128<(KB)*128 +  16>(_lb +  16, gBrow);                              \
      async_b128<(KB)*128 +  32>(_lb +  32, gBrow);                              \
      async_b128<(KB)*128 +  48>(_lb +  48, gBrow);                              \
      async_b128<(KB)*128 +  64>(_lb +  64, gBrow);                              \
      async_b128<(KB)*128 +  80>(_lb +  80, gBrow);                              \
      async_b128<(KB)*128 +  96>(_lb +  96, gBrow);                              \
      async_b128<(KB)*128 + 112>(_lb + 112, gBrow);                              \
      async_b128<(KB)*128      >(_la,        gArow);                             \
    } while (0)

  STAGE(0, 0);
#pragma unroll
  for (int kb = 0; kb < LDIM / KBLK; ++kb) {
    const int buf = kb & 1;
    if (kb < LDIM / KBLK - 1) {
      switch (kb + 1) {      // compile-time staging (loop is fully unrolled)
        case 1: STAGE(1, 1); break;
        case 2: STAGE(2, 0); break;
        case 3: STAGE(3, 1); break;
        case 4: STAGE(4, 0); break;
        case 5: STAGE(5, 1); break;
        case 6: STAGE(6, 0); break;
        case 7: STAGE(7, 1); break;
      }
      wait_async9();
    } else {
      wait_async0();
    }
    __syncthreads();                       // slab `buf` fully in LDS for all waves

    const u16* aP  = aRow  + buf * bufA;
    const u16* b0P = bRow0 + buf * bufB;
    const u16* b1P = bRow1 + buf * bufB;
    const u16* b2P = bRow2 + buf * bufB;
    const u16* b3P = bRow3 + buf * bufB;
#pragma unroll
    for (int ks = 0; ks < KBLK; ks += 32) {
      v16bf af = load_fragA(aP, ks, hi);
      v16bf b0 = load_fragB(b0P, ks, hi);
      v16bf b1 = load_fragB(b1P, ks, hi);
      v16bf b2 = load_fragB(b2P, ks, hi);
      v16bf b3 = load_fragB(b3P, ks, hi);
      WMMA_BF16(acc0, af, b0);
      WMMA_BF16(acc1, af, b1);
      WMMA_BF16(acc2, af, b2);
      WMMA_BF16(acc3, af, b3);
    }
    __syncthreads();                       // everyone done reading before restage
  }
  #undef STAGE

  const int mrow0 = mrowB + (w & 1) * 16;
  const int ncol0 = ncolB + (w >> 1) * 64;
#pragma unroll
  for (int r = 0; r < 8; ++r) {
    int row = mrow0 + r + 8 * hi;   // C/D layout: VGPR r holds M = r (+8 for hi lanes)
    epi_store<EPI>(acc0[r], row, ncol0 + 0 * 16 + ln, z, cZ, Cb, Cf, gvec, bvec, maski, smask);
    epi_store<EPI>(acc1[r], row, ncol0 + 1 * 16 + ln, z, cZ, Cb, Cf, gvec, bvec, maski, smask);
    epi_store<EPI>(acc2[r], row, ncol0 + 2 * 16 + ln, z, cZ, Cb, Cf, gvec, bvec, maski, smask);
    epi_store<EPI>(acc3[r], row, ncol0 + 3 * 16 + ln, z, cZ, Cb, Cf, gvec, bvec, maski, smask);
  }
}

// x = item_emb[positives] + pos_emb  -> bf16 row-major [B*L, 512]
__global__ __launch_bounds__(256) void embed_k(
    const int* __restrict__ positives, const float* __restrict__ item_emb,
    const float* __restrict__ pos_emb, u16* __restrict__ X) {
  size_t idx = (size_t)blockIdx.x * 256 + threadIdx.x;   // B*L*H/4 threads
  int c4     = (int)(idx & 127);
  size_t row = idx >> 7;
  int l      = (int)(row & 511);
  int item   = positives[row];
  float4 ie = *reinterpret_cast<const float4*>(item_emb + (size_t)item * LDIM + c4 * 4);
  float4 pe = *reinterpret_cast<const float4*>(pos_emb + (size_t)l * LDIM + c4 * 4);
  uint2 o;
  o.x = (unsigned)f2bf(ie.x + pe.x) | ((unsigned)f2bf(ie.y + pe.y) << 16);
  o.y = (unsigned)f2bf(ie.z + pe.z) | ((unsigned)f2bf(ie.w + pe.w) << 16);
  *reinterpret_cast<uint2*>(X + idx * 4) = o;
}

// smask = sigmoid((log(w/(1-w)) + gumbel) / 0.2)
__global__ __launch_bounds__(256) void smask_k(
    const float* __restrict__ w, const float* __restrict__ g, float* __restrict__ out) {
  int i = blockIdx.x * 256 + threadIdx.x;                // 512*512 threads
  float ww = w[i];
  float x = (__logf(ww / (1.0f - ww)) + g[i]) * 5.0f;
  out[i] = 1.0f / (1.0f + __expf(-x));
}

// f32 -> bf16 weight conversion (vectorized by 4)
__global__ __launch_bounds__(256) void cvt_k(
    const float* __restrict__ src, u16* __restrict__ dst, int n4) {
  int i = blockIdx.x * 256 + threadIdx.x;
  if (i < n4) {
    float4 f = reinterpret_cast<const float4*>(src)[i];
    uint2 o;
    o.x = (unsigned)f2bf(f.x) | ((unsigned)f2bf(f.y) << 16);
    o.y = (unsigned)f2bf(f.z) | ((unsigned)f2bf(f.w) << 16);
    *reinterpret_cast<uint2*>(dst + (size_t)i * 4) = o;
  }
}

extern "C" void kernel_launch(void* const* d_in, const int* in_sizes, int n_in,
                              void* d_out, int out_size, void* d_ws, size_t ws_size,
                              hipStream_t stream) {
  const int*   positives = (const int*)d_in[0];
  const int*   maski     = (const int*)d_in[1];
  const float* item_emb  = (const float*)d_in[2];
  const float* pos_emb   = (const float*)d_in[3];
  const float* Wz        = (const float*)d_in[4];
  const float* Wv        = (const float*)d_in[5];
  const float* Wq        = (const float*)d_in[6];
  const float* Wk        = (const float*)d_in[7];
  const float* gq        = (const float*)d_in[8];
  const float* bq        = (const float*)d_in[9];
  const float* gk        = (const float*)d_in[10];
  const float* bk        = (const float*)d_in[11];
  const float* sparse_w  = (const float*)d_in[12];
  const float* gumbel    = (const float*)d_in[13];
  float* outp = (float*)d_out;

  char* ws = (char*)d_ws;
  u16*   Xb  = (u16*)(ws + 0);            // 33.5 MB  [B*L,512]   (reused as Aattn)
  u16*   Zb  = (u16*)(ws + 33554432);     // 33.5 MB
  u16*   Vt  = (u16*)(ws + 67108864);     // 33.5 MB  [B][H][L]
  u16*   Qb  = (u16*)(ws + 100663296);    // 33.5 MB
  u16*   Kb  = (u16*)(ws + 134217728);    // 33.5 MB
  float* sm  = (float*)(ws + 167772160);  // 1 MB
  u16*   Wzb = (u16*)(ws + 168820736);
  u16*   Wvb = (u16*)(ws + 169345024);
  u16*   Wqb = (u16*)(ws + 169869312);
  u16*   Wkb = (u16*)(ws + 170393600);

  dim3 blk(256);

  cvt_k<<<256, blk, 0, stream>>>(Wz, Wzb, MAT / 4);
  cvt_k<<<256, blk, 0, stream>>>(Wv, Wvb, MAT / 4);
  cvt_k<<<256, blk, 0, stream>>>(Wq, Wqb, MAT / 4);
  cvt_k<<<256, blk, 0, stream>>>(Wk, Wkb, MAT / 4);
  smask_k<<<1024, blk, 0, stream>>>(sparse_w, gumbel, sm);
  embed_k<<<16384, blk, 0, stream>>>(positives, item_emb, pos_emb, Xb);

  // Z = silu(X @ Wz^T)                   M=32768 N=512
  gemm_k<0><<<dim3(2, 1024, 1), blk, 0, stream>>>(Xb, Wzb, 0, 0, 0, Zb, nullptr,
                                                  nullptr, nullptr, nullptr, nullptr);
  // Vt = silu(Wv @ X^T)                  M=512   N=32768  (stores v transposed per batch)
  gemm_k<1><<<dim3(128, 16, 1), blk, 0, stream>>>(Wvb, Xb, 0, 0, 0, Vt, nullptr,
                                                  nullptr, nullptr, nullptr, nullptr);
  // Q = (Z @ Wq^T) * gq + bq
  gemm_k<2><<<dim3(2, 1024, 1), blk, 0, stream>>>(Zb, Wqb, 0, 0, 0, Qb, nullptr,
                                                  gq, bq, nullptr, nullptr);
  // K = (Z @ Wk^T) * gk + bk
  gemm_k<2><<<dim3(2, 1024, 1), blk, 0, stream>>>(Zb, Wkb, 0, 0, 0, Kb, nullptr,
                                                  gk, bk, nullptr, nullptr);
  // Aattn[b] = relu(smask * mask(Q[b] @ K[b]^T))^2   (per batch; overwrites X scratch)
  gemm_k<3><<<dim3(2, 16, BDIM), blk, 0, stream>>>(Qb, Kb, MAT, MAT, MAT, Xb, nullptr,
                                                   nullptr, nullptr, maski, sm);
  // Out[b] = (Aattn[b] @ V[b]) / (L*H)   (V via its transposed layout)
  gemm_k<4><<<dim3(2, 16, BDIM), blk, 0, stream>>>(Xb, Vt, MAT, MAT, MAT, nullptr, outp,
                                                   nullptr, nullptr, nullptr, nullptr);
}
